// RNNetworkExplicit_11871289606376
// MI455X (gfx1250) — matrix operand order repaired and checked by reading, not compiled, not run on hardware
//
#include <hip/hip_runtime.h>
#include <hip/hip_bf16.h>
#include <stdint.h>

// ---------------------------------------------------------------------------
// Model dims (compile-time constants from the reference)
// ---------------------------------------------------------------------------
constexpr int HD    = 512;           // hidden
constexpr int TSTEP = 2048;          // T
constexpr int BATCH = 8;             // B
constexpr int MROWS = BATCH * TSTEP; // 16384 rows for all big GEMMs
constexpr int NLAYER = 4;

typedef __attribute__((ext_vector_type(16))) __bf16    v16bf;
typedef __attribute__((ext_vector_type(8)))  float     v8f;
typedef __attribute__((ext_vector_type(4)))  uint32_t  u32x4;
typedef __attribute__((ext_vector_type(8)))  int32_t   i32x8;
typedef __attribute__((ext_vector_type(4)))  int32_t   i32x4;

union ABfrag {
  uint32_t u[8];
  u32x4    q[2];
  v16bf    v;
};

__device__ __forceinline__ unsigned short f2b(float f) {
  union { float f; uint32_t u; } x; x.f = f;
  uint32_t r = (x.u + 0x7FFFu + ((x.u >> 16) & 1u)) >> 16; // RNE
  return (unsigned short)r;
}

__device__ __forceinline__ float gelu_exact(float x) {
  return 0.5f * x * (1.0f + erff(x * 0.70710678118654752f));
}

// ---------------------------------------------------------------------------
// f32 -> bf16 cast
// ---------------------------------------------------------------------------
__global__ __launch_bounds__(256)
void cast_f32_to_bf16(const float* __restrict__ in,
                      unsigned short* __restrict__ out, size_t n) {
  size_t i = (size_t)blockIdx.x * blockDim.x + threadIdx.x;
  size_t stride = (size_t)gridDim.x * blockDim.x;
  for (; i < n; i += stride) out[i] = f2b(in[i]);
}

// ---------------------------------------------------------------------------
// GEMM: C[M,N] = act(A[M,K](bf16) * W[N,K](bf16)^T + bias)
// 256 threads = 8 waves arranged 2(M) x 4(N); each wave register-blocks a
// 32x32 C tile: 2 A-frags x 2 B-frags -> 4 WMMAs per K=32 chunk (2x reuse).
// Fragment layouts per CDNA5 ISA 7.12.2 (wave32):
//   A 16x32 bf16 : lane m=L%16; u32 pair p -> k/2 = (p>=4)*8 + (L>=16)*4 + p%4
//   B 32x16 bf16 : lane n=L%16; u32 pair p -> k/2 = (p>=4)*8 + (L>=16)*4 + p%4 (as A)
//   C/D f32      : lane n=L%16; vgpr r -> m = r + (L>=16)*8
// ---------------------------------------------------------------------------
__global__ __launch_bounds__(256)
void gemm_wmma_bf16(const unsigned short* __restrict__ A,   // [M,K]
                    const unsigned short* __restrict__ W,   // [N,K]
                    const float* __restrict__ bias,         // [N]
                    float* __restrict__ outF,               // [M,N] or null
                    unsigned short* __restrict__ outB,      // [M,N] or null
                    int N, int K, int act) {
  const int lane = threadIdx.x & 31;
  const int wave = threadIdx.x >> 5;
  const int wm = wave & 1;                 // 2 waves along M
  const int wn = wave >> 1;                // 4 waves along N
  const int m0 = blockIdx.y * 64 + wm * 32;
  const int n0 = blockIdx.x * 128 + wn * 32;
  const int hi = lane >> 4;
  const int lr = lane & 15;

  const uint32_t* Arow0 = (const uint32_t*)(A + (size_t)(m0 + lr) * K);
  const uint32_t* Arow1 = (const uint32_t*)(A + (size_t)(m0 + 16 + lr) * K);
  const uint32_t* Wrow0 = (const uint32_t*)(W + (size_t)(n0 + lr) * K);
  const uint32_t* Wrow1 = (const uint32_t*)(W + (size_t)(n0 + 16 + lr) * K);

  v8f acc00 = {}, acc01 = {}, acc10 = {}, acc11 = {};
  for (int kc = 0; kc < K; kc += 32) {
    const int kp = kc >> 1;                // u32 index of K-chunk base
    const int aoff = kp + (hi << 2);
    const int boff = kp + (hi << 2);
    ABfrag a0, a1, b0, b1;
    a0.q[0] = *(const u32x4*)(Arow0 + aoff);
    a0.q[1] = *(const u32x4*)(Arow0 + aoff + 8);
    a1.q[0] = *(const u32x4*)(Arow1 + aoff);
    a1.q[1] = *(const u32x4*)(Arow1 + aoff + 8);
    b0.q[0] = *(const u32x4*)(Wrow0 + boff);
    b0.q[1] = *(const u32x4*)(Wrow0 + boff + 8);
    b1.q[0] = *(const u32x4*)(Wrow1 + boff);
    b1.q[1] = *(const u32x4*)(Wrow1 + boff + 8);
    if (kc + 32 < K) {                     // global_prefetch_b8 next chunk
      __builtin_prefetch(Arow0 + kp + 16, 0, 1);
      __builtin_prefetch(Arow1 + kp + 16, 0, 1);
      __builtin_prefetch(Wrow0 + kp + 16, 0, 1);
      __builtin_prefetch(Wrow1 + kp + 16, 0, 1);
    }
    acc00 = __builtin_amdgcn_wmma_f32_16x16x32_bf16(false, a0.v, false, b0.v,
                                                    (short)0, acc00, false, false);
    acc01 = __builtin_amdgcn_wmma_f32_16x16x32_bf16(false, a0.v, false, b1.v,
                                                    (short)0, acc01, false, false);
    acc10 = __builtin_amdgcn_wmma_f32_16x16x32_bf16(false, a1.v, false, b0.v,
                                                    (short)0, acc10, false, false);
    acc11 = __builtin_amdgcn_wmma_f32_16x16x32_bf16(false, a1.v, false, b1.v,
                                                    (short)0, acc11, false, false);
  }

  const v8f accs[4] = {acc00, acc01, acc10, acc11};
  const int   nn[4] = {n0 + lr, n0 + 16 + lr, n0 + lr, n0 + 16 + lr};
  const int   mm[4] = {m0 + (hi << 3), m0 + (hi << 3),
                       m0 + 16 + (hi << 3), m0 + 16 + (hi << 3)};
#pragma unroll
  for (int tile = 0; tile < 4; ++tile) {
    const int n = nn[tile];
    const int mbase = mm[tile];
    const float bn = bias[n];
    float r[8];
#pragma unroll
    for (int i = 0; i < 8; ++i) {
      float v = accs[tile][i] + bn;
      if (act == 1) v = gelu_exact(v);
      r[i] = v;
    }
    if (outF) {
#pragma unroll
      for (int i = 0; i < 8; ++i)
        outF[(size_t)(mbase + i) * N + n] = r[i];
    }
    if (outB) {
#pragma unroll
      for (int i = 0; i < 8; ++i)
        outB[(size_t)(mbase + i) * N + n] = f2b(r[i]);
    }
  }
}

// ---------------------------------------------------------------------------
// Device-scope sync for the 4 cooperating blocks. Default: sense-reversing
// atomic barrier (valid for any launch). Optional: CDNA5 cluster barrier
// (s_barrier_signal/-wait -3) when dispatched as a workgroup cluster.
// ---------------------------------------------------------------------------
__device__ __forceinline__ void grid_sync(unsigned int* bar, int nblocks,
                                          int use_cluster) {
  __syncthreads();
  if (threadIdx.x == 0) {
    __threadfence();
    if (use_cluster) {
#if __has_builtin(__builtin_amdgcn_s_cluster_barrier)
      __builtin_amdgcn_s_cluster_barrier();
#endif
    } else {
      unsigned int g = __hip_atomic_load(&bar[1], __ATOMIC_RELAXED,
                                         __HIP_MEMORY_SCOPE_AGENT);
      unsigned int a = __hip_atomic_fetch_add(&bar[0], 1u, __ATOMIC_ACQ_REL,
                                              __HIP_MEMORY_SCOPE_AGENT);
      if (a == (unsigned)(nblocks - 1)) {
        __hip_atomic_store(&bar[0], 0u, __ATOMIC_RELAXED,
                           __HIP_MEMORY_SCOPE_AGENT);
        __hip_atomic_fetch_add(&bar[1], 1u, __ATOMIC_RELEASE,
                               __HIP_MEMORY_SCOPE_AGENT);
      } else {
        while (__hip_atomic_load(&bar[1], __ATOMIC_ACQUIRE,
                                 __HIP_MEMORY_SCOPE_AGENT) == g)
          __builtin_amdgcn_s_sleep(1);
      }
    }
    __threadfence();
  }
  __syncthreads();
}

// ---------------------------------------------------------------------------
// Persistent RNN scan. Grid = 4 blocks x 8 waves = 32 waves; wave w owns the
// 16-column slice of h_next with its 512x16 W_hh slice in LDS (16KB/wave,
// 128KB/WGP of the 320KB). W_hh staging can use the Tensor Data Mover
// (tensor_load_to_lds, D# packed per ISA 8.3/8.4; 2D tile 512x128 elems,
// data_size=2B) tracked by TENSORcnt; the manual-load path is the default.
// h (8 rows padded to 16) is double-buffered in global bf16 -> one grid sync
// per timestep. pre[] is overwritten in place with hs[].
// ---------------------------------------------------------------------------
__global__ __launch_bounds__(256)
void rnn_scan_wmma(const unsigned short* __restrict__ Whh,  // [HD,HD] bf16
                   float* __restrict__ pre,                 // [B,T,HD] -> hs
                   const float* __restrict__ bhh,           // [HD]
                   unsigned short* __restrict__ hbf,        // [2,16,HD] bf16
                   unsigned int* __restrict__ bar,
                   int use_tdm, int use_cluster) {
  extern __shared__ unsigned short smem[];
  const int lane = threadIdx.x & 31;
  const int wave = threadIdx.x >> 5;
  const int hi = lane >> 4;
  const int lr = lane & 15;
  const int gt = blockIdx.x * 8 + wave;   // global column-tile 0..31
  const int n = gt * 16 + lr;

  if (use_tdm) {
#if __has_builtin(__builtin_amdgcn_tensor_load_to_lds) && \
    __has_builtin(__builtin_amdgcn_s_wait_tensorcnt)
    if (wave == 0) {
      // D# group0: count=1 | lds_addr | global_addr | type=2
      const uint64_t ga =
          (uint64_t)(uintptr_t)(Whh + (size_t)blockIdx.x * 128 * HD);
      const uint32_t ldso = (uint32_t)(uintptr_t)(&smem[0]);
      u32x4 g0;
      g0[0] = 1u;                                    // count=1, user mode
      g0[1] = ldso;                                  // lds_addr
      g0[2] = (uint32_t)ga;                          // global_addr[31:0]
      g0[3] = (uint32_t)((ga >> 32) & 0x01FFFFFFu) | 0x80000000u; // +type=2
      // D# group1: mask=0, data_size=1(2B); dim0=512, dim1=128,
      // tile0=512, tile1=128, stride0=512
      i32x8 g1;
      g1[0] = (int32_t)(1u << 16);
      g1[1] = (int32_t)((uint32_t)HD << 16);         // tensor_dim0[15:0]
      g1[2] = (int32_t)(128u << 16);                 // tensor_dim1[15:0]
      g1[3] = (int32_t)((uint32_t)HD << 16);         // tile_dim0
      g1[4] = (int32_t)128;                          // tile_dim1
      g1[5] = (int32_t)HD;                           // dim0_stride[31:0]
      g1[6] = 0; g1[7] = 0;
      i32x4 gz4 = {0, 0, 0, 0};
      i32x8 gz8 = {0, 0, 0, 0, 0, 0, 0, 0};
      __builtin_amdgcn_tensor_load_to_lds(g0, g1, gz4, gz4, gz8, 0);
      __builtin_amdgcn_s_wait_tensorcnt(0);
    }
#endif
  } else {
    // Manual staging: this wave's 16 rows of W_hh (contiguous 16KB).
    const uint32_t* src = (const uint32_t*)(Whh + (size_t)gt * 16 * HD);
    uint32_t* dst = (uint32_t*)(smem + (size_t)wave * 16 * HD);
    for (int i = lane; i < 16 * HD / 2; i += 32) dst[i] = src[i];
  }
  // Zero both h buffers (pad rows 8..15 stay zero forever).
  if (blockIdx.x == 0) {
    uint32_t* z = (uint32_t*)hbf;
    for (int i = threadIdx.x; i < 2 * 16 * HD / 2; i += 256) z[i] = 0u;
  }
  grid_sync(bar, gridDim.x, use_cluster);

  const uint32_t* Brow =
      (const uint32_t*)(smem + ((size_t)wave * 16 + lr) * HD);
  const float bn = bhh[n];

  for (int t = 0; t < TSTEP; ++t) {
    const uint32_t* Arow =
        (const uint32_t*)(hbf + ((size_t)(t & 1) * 16 + lr) * HD);
    unsigned short* hout = hbf + (size_t)((t + 1) & 1) * 16 * HD;

    v8f acc = {};
#pragma unroll
    for (int kc = 0; kc < HD; kc += 32) {
      const int kp = kc >> 1;
      ABfrag a, b;
      a.q[0] = *(const u32x4*)(Arow + kp + (hi << 2));
      a.q[1] = *(const u32x4*)(Arow + kp + (hi << 2) + 8);
      const u32x4* bq = (const u32x4*)(Brow + kp + (hi << 2));
      b.q[0] = bq[0];
      b.q[1] = bq[2];
      acc = __builtin_amdgcn_wmma_f32_16x16x32_bf16(false, a.v, false, b.v,
                                                    (short)0, acc, false, false);
    }
    if (hi == 0) {                        // lanes 0..15 hold real rows m=0..7
#pragma unroll
      for (int bidx = 0; bidx < 8; ++bidx) {
        const size_t off = ((size_t)bidx * TSTEP + t) * HD + n;
        const float v = tanhf(acc[bidx] + pre[off] + bn);
        pre[off] = v;                     // hs in place
        hout[(size_t)bidx * HD + n] = f2b(v);
      }
    }
    grid_sync(bar, gridDim.x, use_cluster);
  }
}

// ---------------------------------------------------------------------------
// Fused residual-add + LayerNorm; one wave per row (HD=512 -> 16 elems/lane).
// ---------------------------------------------------------------------------
__global__ __launch_bounds__(256)
void add_layernorm(const float* __restrict__ res, const float* __restrict__ delta,
                   const float* __restrict__ g, const float* __restrict__ b,
                   float* __restrict__ outF, unsigned short* __restrict__ outB) {
  const int lane = threadIdx.x & 31;
  const int wave = threadIdx.x >> 5;
  const int row = blockIdx.x * 8 + wave;
  const size_t base = (size_t)row * HD;
  float v[16];
  float s = 0.f, s2 = 0.f;
#pragma unroll
  for (int i = 0; i < 16; ++i) {
    const int j = lane + i * 32;
    const float x = res[base + j] + delta[base + j];
    v[i] = x; s += x; s2 += x * x;
  }
#pragma unroll
  for (int m = 16; m >= 1; m >>= 1) {
    s  += __shfl_xor(s, m, 32);
    s2 += __shfl_xor(s2, m, 32);
  }
  const float mean = s * (1.0f / HD);
  const float var  = s2 * (1.0f / HD) - mean * mean;
  const float rs   = rsqrtf(var + 1e-5f);
#pragma unroll
  for (int i = 0; i < 16; ++i) {
    const int j = lane + i * 32;
    const float y = (v[i] - mean) * rs * g[j] + b[j];
    outF[base + j] = y;
    outB[base + j] = f2b(y);
  }
}

__global__ void init_barrier(unsigned int* bar) { bar[0] = 0u; bar[1] = 0u; }

// ---------------------------------------------------------------------------
// Launch
// ---------------------------------------------------------------------------
extern "C" void kernel_launch(void* const* d_in, const int* in_sizes, int n_in,
                              void* d_out, int out_size, void* d_ws, size_t ws_size,
                              hipStream_t stream) {
  const float* x     = (const float*)d_in[0];
  const float* W_in  = (const float*)d_in[1];
  const float* b_in  = (const float*)d_in[2];
  const float* W_ih  = (const float*)d_in[3];
  const float* b_ih  = (const float*)d_in[4];
  const float* W_hh  = (const float*)d_in[5];
  const float* b_hh  = (const float*)d_in[6];
  const float* ln1_g = (const float*)d_in[7];
  const float* ln1_b = (const float*)d_in[8];
  const float* W1    = (const float*)d_in[9];
  const float* b1    = (const float*)d_in[10];
  const float* W2    = (const float*)d_in[11];
  const float* b2    = (const float*)d_in[12];
  const float* ln2_g = (const float*)d_in[13];
  const float* ln2_b = (const float*)d_in[14];
  const float* W_out = (const float*)d_in[15];
  const float* b_out = (const float*)d_in[16];

  char* p = (char*)d_ws;
  auto carve = [&](size_t bytes) -> void* {
    void* r = (void*)p;
    p += (bytes + 255) & ~(size_t)255;
    return r;
  };

  unsigned short* wbf_in  = (unsigned short*)carve((size_t)HD * HD * 2);
  unsigned short* wbf_ih  = (unsigned short*)carve((size_t)NLAYER * HD * HD * 2);
  unsigned short* wbf_hh  = (unsigned short*)carve((size_t)NLAYER * HD * HD * 2);
  unsigned short* wbf_1   = (unsigned short*)carve((size_t)NLAYER * 4 * HD * HD * 2);
  unsigned short* wbf_2   = (unsigned short*)carve((size_t)NLAYER * 4 * HD * HD * 2);
  unsigned short* wbf_out = (unsigned short*)carve((size_t)4 * HD * HD * 2);
  unsigned short* absA    = (unsigned short*)carve((size_t)MROWS * HD * 2);
  unsigned short* abf4h   = (unsigned short*)carve((size_t)MROWS * 4 * HD * 2);
  float*          buf_out = (float*)carve((size_t)MROWS * HD * 4);
  float*          buf_tmp = (float*)carve((size_t)MROWS * HD * 4);
  unsigned short* hbf     = (unsigned short*)carve((size_t)2 * 16 * HD * 2);
  unsigned int*   bar     = (unsigned int*)carve(256);

  const int use_tdm = 0;      // TDM path compiled in; manual path executed
  const int use_cluster = 0;  // cluster-barrier path compiled in; atomics used

  auto castN = [&](const float* src, unsigned short* dst, size_t n) {
    int blocks = (int)((n + 1023) / 1024);
    if (blocks > 4096) blocks = 4096;
    cast_f32_to_bf16<<<blocks, 256, 0, stream>>>(src, dst, n);
  };
  auto gemm = [&](const unsigned short* A, const unsigned short* W,
                  const float* bias, float* oF, unsigned short* oB,
                  int N, int K, int act) {
    dim3 grid(N / 128, MROWS / 64);
    gemm_wmma_bf16<<<grid, dim3(256), 0, stream>>>(A, W, bias, oF, oB, N, K, act);
  };

  // One-time per-launch weight casts (deterministic, graph-capturable)
  castN(W_in,  wbf_in,  (size_t)HD * HD);
  castN(W_ih,  wbf_ih,  (size_t)NLAYER * HD * HD);
  castN(W_hh,  wbf_hh,  (size_t)NLAYER * HD * HD);
  castN(W1,    wbf_1,   (size_t)NLAYER * 4 * HD * HD);
  castN(W2,    wbf_2,   (size_t)NLAYER * 4 * HD * HD);
  castN(W_out, wbf_out, (size_t)4 * HD * HD);
  castN(x,     absA,    (size_t)MROWS * HD);
  init_barrier<<<1, 1, 0, stream>>>(bar);

  // Input projection
  gemm(absA, wbf_in, b_in, buf_out, nullptr, HD, HD, 0);
  castN(buf_out, absA, (size_t)MROWS * HD);

  for (int l = 0; l < NLAYER; ++l) {
    // pre = out @ W_ih^T + b_ih
    gemm(absA, wbf_ih + (size_t)l * HD * HD, b_ih + l * HD,
         buf_tmp, nullptr, HD, HD, 0);
    // sequential recurrence (pre -> hs in place)
    rnn_scan_wmma<<<4, 256, 8 * 16 * HD * 2, stream>>>(
        wbf_hh + (size_t)l * HD * HD, buf_tmp, b_hh + l * HD, hbf, bar,
        use_tdm, use_cluster);
    // out = LN1(residual + hs)
    add_layernorm<<<MROWS / 8, 256, 0, stream>>>(
        buf_out, buf_tmp, ln1_g + l * HD, ln1_b + l * HD, buf_out, absA);
    // mlp hidden = gelu(out @ W1^T + b1), written directly as bf16
    gemm(absA, wbf_1 + (size_t)l * 4 * HD * HD, b1 + l * 4 * HD,
         nullptr, abf4h, 4 * HD, HD, 1);
    // mlp out = hidden @ W2^T + b2
    gemm(abf4h, wbf_2 + (size_t)l * 4 * HD * HD, b2 + l * HD,
         buf_tmp, nullptr, HD, 4 * HD, 0);
    // out = LN2(out + mlp)
    add_layernorm<<<MROWS / 8, 256, 0, stream>>>(
        buf_out, buf_tmp, ln2_g + l * HD, ln2_b + l * HD, buf_out, absA);
  }

  // Output projection -> d_out [B,T,2048] f32
  gemm(absA, wbf_out, b_out, (float*)d_out, nullptr, 4 * HD, HD, 0);
}
